// DepthDeltaNetDecoderLayer_55929064129107
// MI455X (gfx1250) — matrix operand order repaired and checked by reading, not compiled, not run on hardware
//
#include <hip/hip_runtime.h>
#include <math.h>

typedef __attribute__((ext_vector_type(16))) _Float16 v16h;
typedef __attribute__((ext_vector_type(8)))  _Float16 v8h;
typedef __attribute__((ext_vector_type(8)))  float    v8f;
typedef __attribute__((ext_vector_type(4)))  float    v4f;

#define BM 128
#define BN 128
#define BK 32

__device__ __forceinline__ float sigmoidf_(float x) { return 1.0f / (1.0f + __expf(-x)); }
__device__ __forceinline__ float siluf_(float x)    { return x / (1.0f + __expf(-x)); }

// ---------------------------------------------------------------------------
// One-time weight pack: W f32 [K][N] row-major -> Bp f16 [N][K] (k-contiguous).
// 32x32 tiles through LDS so both global read and write are coalesced.
// ---------------------------------------------------------------------------
__global__ void __launch_bounds__(256)
pack_w_kernel(const float* __restrict__ W, _Float16* __restrict__ Bp, int K, int N)
{
    __shared__ float tile[32][33];
    const int n0 = blockIdx.x * 32;
    const int k0 = blockIdx.y * 32;
    const int tx = threadIdx.x & 31;
    const int ty = threadIdx.x >> 5;      // 0..7
#pragma unroll
    for (int i = 0; i < 4; ++i) {
        int k = k0 + ty + i * 8;
        int n = n0 + tx;
        tile[ty + i * 8][tx] = (n < N) ? W[(size_t)k * N + n] : 0.0f;
    }
    __syncthreads();
#pragma unroll
    for (int i = 0; i < 4; ++i) {
        int n = n0 + ty + i * 8;
        if (n < N)
            Bp[(size_t)n * K + k0 + tx] = (_Float16)tile[tx][ty + i * 8];
    }
}

// ---------------------------------------------------------------------------
// WMMA GEMM: C[M,N] = epilogue(A_f16[M,K] @ Bp_f16[N,K]^T) (+R)
// act: 0=none 1=sigmoid 2=silu ; out_half: store _Float16
// 256 threads = 8 wave32; block tile 128x128x32; wave tile 64x32
// (4x2 v_wmma_f32_16x16x32_f16). Double-buffered LDS + register staging:
// one __syncthreads per K-tile. M must be a multiple of 128.
// NGUARD=false additionally requires N to be a multiple of 128 (branch-free).
// ---------------------------------------------------------------------------
template<bool NGUARD>
__global__ void __launch_bounds__(256)
gemm_wmma(const _Float16* __restrict__ A, const _Float16* __restrict__ Bp,
          void* __restrict__ Cout, const float* __restrict__ R,
          int M, int N, int K, int act, int out_half)
{
    __shared__ _Float16 ldsA[2][BM * BK];   // [m][k] row-major, 8 KB each
    __shared__ _Float16 ldsB[2][BN * BK];   // [n][k] col-major, 8 KB each

    const int tid   = threadIdx.x;
    const int lane  = tid & 31;
    const int wid   = tid >> 5;
    const int waveM = wid >> 2;   // 0..1 -> 64-row slab
    const int waveN = wid & 3;    // 0..3 -> 32-col slab
    const int bm    = blockIdx.y * BM;
    const int bn    = blockIdx.x * BN;

    // Staging: 512 v8h chunks per tile; each thread owns rows srow, srow+64.
    const int srow = tid >> 2;          // 0..63
    const int sco  = (tid & 3) * 8;     // 0,8,16,24

    const _Float16* Ag0 = A  + (size_t)(bm + srow)      * K + sco;
    const _Float16* Ag1 = A  + (size_t)(bm + srow + 64) * K + sco;
    const _Float16* Bg0 = Bp + (size_t)(bn + srow)      * K + sco;
    const _Float16* Bg1 = Bp + (size_t)(bn + srow + 64) * K + sco;
    const bool bok0 = !NGUARD || (bn + srow)      < N;
    const bool bok1 = !NGUARD || (bn + srow + 64) < N;

    v8h zero8;
#pragma unroll
    for (int j = 0; j < 8; ++j) zero8[j] = (_Float16)0.0f;

    v8h a0 = *(const v8h*)(Ag0);
    v8h a1 = *(const v8h*)(Ag1);
    v8h b0 = zero8, b1 = zero8;
    if (bok0) b0 = *(const v8h*)(Bg0);
    if (bok1) b1 = *(const v8h*)(Bg1);

    v8f acc[4][2];
#pragma unroll
    for (int ms = 0; ms < 4; ++ms)
#pragma unroll
        for (int ns = 0; ns < 2; ++ns)
#pragma unroll
            for (int j = 0; j < 8; ++j) acc[ms][ns][j] = 0.0f;

    const int r16 = lane & 15;
    const int g   = lane >> 4;
    const int nk  = K / BK;

    for (int kt = 0; kt < nk; ++kt) {
        const int buf = kt & 1;
        *(v8h*)(ldsA[buf] + srow * BK + sco)        = a0;
        *(v8h*)(ldsA[buf] + (srow + 64) * BK + sco) = a1;
        *(v8h*)(ldsB[buf] + srow * BK + sco)        = b0;
        *(v8h*)(ldsB[buf] + (srow + 64) * BK + sco) = b1;
        if (kt + 1 < nk) {                 // stage next tile while this computes
            const int ko = (kt + 1) * BK;
            a0 = *(const v8h*)(Ag0 + ko);
            a1 = *(const v8h*)(Ag1 + ko);
            if (bok0) b0 = *(const v8h*)(Bg0 + ko);
            if (bok1) b1 = *(const v8h*)(Bg1 + ko);
        }
        __syncthreads();                   // single barrier: ping-pong buffers

        // Fragments per ISA 16-bit layouts.
        v16h afrag[4];
#pragma unroll
        for (int ms = 0; ms < 4; ++ms) {
            const _Float16* p = ldsA[buf] + (waveM * 64 + ms * 16 + r16) * BK;
            v8h lo = *(const v8h*)(p + g * 8);        // K = g*8 ..
            v8h hi = *(const v8h*)(p + 16 + g * 8);   // K = 16+g*8 ..
            v16h f;
#pragma unroll
            for (int j = 0; j < 8; ++j) { f[j] = lo[j]; f[j + 8] = hi[j]; }
            afrag[ms] = f;
        }
        v16h bfrag[2];
#pragma unroll
        for (int ns = 0; ns < 2; ++ns) {
            const _Float16* p = ldsB[buf] + (waveN * 32 + ns * 16 + r16) * BK;
            v8h lo = *(const v8h*)(p + g * 8);        // K = g*8 ..
            v8h hi = *(const v8h*)(p + 16 + g * 8);   // K = 16+g*8 ..
            v16h f;
#pragma unroll
            for (int j = 0; j < 8; ++j) { f[j] = lo[j]; f[j + 8] = hi[j]; }
            bfrag[ns] = f;
        }
#pragma unroll
        for (int ms = 0; ms < 4; ++ms)
#pragma unroll
            for (int ns = 0; ns < 2; ++ns)
                acc[ms][ns] = __builtin_amdgcn_wmma_f32_16x16x32_f16(
                    false, afrag[ms], false, bfrag[ns],
                    (short)0, acc[ms][ns], false, false);
    }

    // Epilogue: C/D layout = lane%16 -> N, VGPR r -> M = (lane/16)*8 + r.
    const int cn = lane & 15;
    const int cg = lane >> 4;
#pragma unroll
    for (int ms = 0; ms < 4; ++ms) {
#pragma unroll
        for (int ns = 0; ns < 2; ++ns) {
            int n = bn + waveN * 32 + ns * 16 + cn;
            if (NGUARD && n >= N) continue;
#pragma unroll
            for (int r = 0; r < 8; ++r) {
                int m = bm + waveM * 64 + ms * 16 + cg * 8 + r;
                float v = acc[ms][ns][r];
                if (act == 1)      v = sigmoidf_(v);
                else if (act == 2) v = siluf_(v);
                if (R) v += R[(size_t)m * N + n];
                if (out_half) ((_Float16*)Cout)[(size_t)m * N + n] = (_Float16)v;
                else          ((float*)Cout)[(size_t)m * N + n]    = v;
            }
        }
    }
}

// ---------------------------------------------------------------------------
// RMSNorm over D=1024 + cast to f16. One block per row, 4 elems/thread.
// ---------------------------------------------------------------------------
__global__ void __launch_bounds__(256)
rmsnorm_cast_kernel(const float* __restrict__ x, const float* __restrict__ w,
                    _Float16* __restrict__ out, int D)
{
    __shared__ float red[256];
    const int row = blockIdx.x;
    const int tid = threadIdx.x;
    const float* xr = x + (size_t)row * D;
    v4f xv = *(const v4f*)(xr + tid * 4);
    red[tid] = xv[0]*xv[0] + xv[1]*xv[1] + xv[2]*xv[2] + xv[3]*xv[3];
    __syncthreads();
    for (int s = 128; s > 0; s >>= 1) {
        if (tid < s) red[tid] += red[tid + s];
        __syncthreads();
    }
    float rms = rsqrtf(red[0] / (float)D + 1e-6f);
#pragma unroll
    for (int j = 0; j < 4; ++j)
        out[(size_t)row * D + tid * 4 + j] = (_Float16)(xv[j] * rms * w[tid * 4 + j]);
}

// ---------------------------------------------------------------------------
// RoPE + L2-normalize q and k in place. One wave32 per (t, h); lane l owns
// dims l and l+32 (same rotation angle); wave32 __shfl_xor reduction.
// ---------------------------------------------------------------------------
__global__ void __launch_bounds__(256)
rope_l2_kernel(float* __restrict__ q, float* __restrict__ k, int T, int H)
{
    int warp = blockIdx.x * 8 + (threadIdx.x >> 5);
    int lane = threadIdx.x & 31;
    int t = warp >> 4;
    int h = warp & 15;
    if (t >= T) return;
    // inv_freq[l] = 10000^(-2l/64) = 10000^(-l/32)
    float ang = (float)t * __powf(10000.0f, -(float)lane * (1.0f / 32.0f));
    float c = __cosf(ang), s = __sinf(ang);
#pragma unroll
    for (int which = 0; which < 2; ++which) {
        float* base = (which ? k : q) + ((size_t)t * H + h) * 64;
        float x1 = base[lane], x2 = base[lane + 32];
        float y1 = x1 * c - x2 * s;   // [:32] : x*cos - x2*sin
        float y2 = x2 * c + x1 * s;   // [32:] : x*cos + x1*sin
        float ss = y1 * y1 + y2 * y2;
        for (int m = 16; m > 0; m >>= 1) ss += __shfl_xor(ss, m, 32);
        float rn = rsqrtf(ss + 1e-12f);
        base[lane]      = y1 * rn;
        base[lane + 32] = y2 * rn;
    }
}

// ---------------------------------------------------------------------------
// Sequential delta-rule scan. One block per head (16 blocks, 256 threads).
// Thread (v = tid&63, kg = tid>>6) owns S[kg*16 .. kg*16+15][v] in registers.
// ---------------------------------------------------------------------------
__global__ void __launch_bounds__(256)
delta_scan_kernel(const float* __restrict__ q, const float* __restrict__ k,
                  const float* __restrict__ v, const float* __restrict__ alpha,
                  const float* __restrict__ beta, float* __restrict__ o,
                  int T, int H)
{
    const int h    = blockIdx.x;
    const int tid  = threadIdx.x;
    const int vcol = tid & 63;
    const int kg   = tid >> 6;
    __shared__ float qs[64], ks[64], vs[64], ds[64];
    __shared__ float po[256], pk[256];
    __shared__ float ab[2];

    float s[16];
#pragma unroll
    for (int i = 0; i < 16; ++i) s[i] = 0.0f;

    for (int t = 0; t < T; ++t) {
        size_t base = ((size_t)t * H + h) * 64;
        if (tid < 64) {
            qs[tid] = q[base + tid];
            ks[tid] = k[base + tid];
            vs[tid] = v[base + tid];
        }
        if (tid == 0) {
            ab[0] = alpha[(size_t)t * H + h];
            ab[1] = beta[(size_t)t * H + h];
        }
        __syncthreads();
        float so = 0.0f, sk = 0.0f;
#pragma unroll
        for (int i = 0; i < 16; ++i) {
            so += s[i] * qs[kg * 16 + i];   // o[v]  = sum_k S[k][v] q[k]
            sk += s[i] * ks[kg * 16 + i];   // Sk[v] = sum_k S[k][v] k[k]
        }
        po[tid] = so; pk[tid] = sk;
        __syncthreads();
        if (tid < 64) {
            float ot  = po[tid] + po[tid + 64] + po[tid + 128] + po[tid + 192];
            float skv = pk[tid] + pk[tid + 64] + pk[tid + 128] + pk[tid + 192];
            o[base + tid] = ot;
            ds[tid] = vs[tid] - skv;        // delta = v - S^T k
        }
        __syncthreads();
        float a = ab[0], b = ab[1];
        float d = ds[vcol];
#pragma unroll
        for (int i = 0; i < 16; ++i)
            s[i] = a * s[i] + b * ks[kg * 16 + i] * d;  // S = a*S + b*k(x)delta
        __syncthreads();
    }
}

// ---------------------------------------------------------------------------
// o -> RMSNorm(o)*o_norm_w * silu(gate), cast to f16. Wave per (t, h).
// ---------------------------------------------------------------------------
__global__ void __launch_bounds__(256)
opost_kernel(const float* __restrict__ o, const float* __restrict__ gate,
             const float* __restrict__ onw, _Float16* __restrict__ oh,
             int T, int H)
{
    int warp = blockIdx.x * 8 + (threadIdx.x >> 5);
    int lane = threadIdx.x & 31;
    int t = warp >> 4;
    int h = warp & 15;
    if (t >= T) return;
    size_t base = ((size_t)t * H + h) * 64;
    float x1 = o[base + lane], x2 = o[base + lane + 32];
    float ss = x1 * x1 + x2 * x2;
    for (int m = 16; m > 0; m >>= 1) ss += __shfl_xor(ss, m, 32);
    float rms = rsqrtf(ss * (1.0f / 64.0f) + 1e-6f);
    float g1 = gate[base + lane], g2 = gate[base + lane + 32];
    oh[base + lane]      = (_Float16)(x1 * rms * onw[lane]      * siluf_(g1));
    oh[base + lane + 32] = (_Float16)(x2 * rms * onw[lane + 32] * siluf_(g2));
}

// ---------------------------------------------------------------------------
// SwiGLU elementwise: mg (already silu'd) *= mu, in place (f16).
// ---------------------------------------------------------------------------
__global__ void __launch_bounds__(256)
mulcast_kernel(_Float16* __restrict__ mg, const _Float16* __restrict__ mu, size_t n)
{
    size_t i = (size_t)blockIdx.x * 256 + threadIdx.x;
    if (i < n) mg[i] = (_Float16)((float)mg[i] * (float)mu[i]);
}

// ---------------------------------------------------------------------------
extern "C" void kernel_launch(void* const* d_in, const int* in_sizes, int n_in,
                              void* d_out, int out_size, void* d_ws, size_t ws_size,
                              hipStream_t stream)
{
    (void)in_sizes; (void)n_in; (void)out_size; (void)ws_size;
    const float* hidden     = (const float*)d_in[0];
    const float* norm1_w    = (const float*)d_in[1];
    const float* q_w        = (const float*)d_in[2];
    const float* k_w        = (const float*)d_in[3];
    const float* v_w        = (const float*)d_in[4];
    const float* g_w        = (const float*)d_in[5];
    const float* b_w        = (const float*)d_in[6];
    const float* gate_w     = (const float*)d_in[7];
    const float* o_norm_w   = (const float*)d_in[8];
    const float* o_w        = (const float*)d_in[9];
    const float* norm2_w    = (const float*)d_in[10];
    const float* mlp_gate_w = (const float*)d_in[11];
    const float* mlp_up_w   = (const float*)d_in[12];
    const float* mlp_down_w = (const float*)d_in[13];

    const int T = 2048, D = 1024, H = 16, FF = 4096;

    char* ws = (char*)d_ws;
    size_t off = 0;
    auto alloc = [&](size_t bytes) -> void* {
        void* p = ws + off;
        off += (bytes + 255) & ~(size_t)255;
        return p;
    };
    // activations
    _Float16* x_h    = (_Float16*)alloc((size_t)T * D * 2);
    float*    qb     = (float*)   alloc((size_t)T * D * 4);
    float*    kb     = (float*)   alloc((size_t)T * D * 4);
    float*    vb     = (float*)   alloc((size_t)T * D * 4);
    float*    gateb  = (float*)   alloc((size_t)T * D * 4);
    float*    alphab = (float*)   alloc((size_t)T * H * 4);
    float*    betab  = (float*)   alloc((size_t)T * H * 4);
    float*    ob     = (float*)   alloc((size_t)T * D * 4);
    _Float16* o_h    = (_Float16*)alloc((size_t)T * D * 2);
    float*    hb     = (float*)   alloc((size_t)T * D * 4);
    _Float16* y_h    = (_Float16*)alloc((size_t)T * D * 2);
    _Float16* mg_h   = (_Float16*)alloc((size_t)T * FF * 2);
    _Float16* mu_h   = (_Float16*)alloc((size_t)T * FF * 2);
    // packed f16 weights, [N][K] k-contiguous
    _Float16* qp  = (_Float16*)alloc((size_t)D * D * 2);
    _Float16* kp  = (_Float16*)alloc((size_t)D * D * 2);
    _Float16* vp  = (_Float16*)alloc((size_t)D * D * 2);
    _Float16* gp  = (_Float16*)alloc((size_t)H * D * 2);
    _Float16* bp  = (_Float16*)alloc((size_t)H * D * 2);
    _Float16* gtp = (_Float16*)alloc((size_t)D * D * 2);
    _Float16* op  = (_Float16*)alloc((size_t)D * D * 2);
    _Float16* mgp = (_Float16*)alloc((size_t)FF * D * 2);
    _Float16* mup = (_Float16*)alloc((size_t)FF * D * 2);
    _Float16* mdp = (_Float16*)alloc((size_t)D * FF * 2);

    dim3 blk(256);
    dim3 pDD(D / 32, D / 32);      // K=D, N=D
    dim3 pDH(1, D / 32);           // K=D, N=16
    dim3 pDF(FF / 32, D / 32);     // K=D, N=FF
    dim3 pFD(D / 32, FF / 32);     // K=FF, N=D

    // 0) pack weights f32 -> f16 [N][K]
    pack_w_kernel<<<pDD, blk, 0, stream>>>(q_w,        qp,  D, D);
    pack_w_kernel<<<pDD, blk, 0, stream>>>(k_w,        kp,  D, D);
    pack_w_kernel<<<pDD, blk, 0, stream>>>(v_w,        vp,  D, D);
    pack_w_kernel<<<pDH, blk, 0, stream>>>(g_w,        gp,  D, H);
    pack_w_kernel<<<pDH, blk, 0, stream>>>(b_w,        bp,  D, H);
    pack_w_kernel<<<pDD, blk, 0, stream>>>(gate_w,     gtp, D, D);
    pack_w_kernel<<<pDD, blk, 0, stream>>>(o_w,        op,  D, D);
    pack_w_kernel<<<pDF, blk, 0, stream>>>(mlp_gate_w, mgp, D, FF);
    pack_w_kernel<<<pDF, blk, 0, stream>>>(mlp_up_w,   mup, D, FF);
    pack_w_kernel<<<pFD, blk, 0, stream>>>(mlp_down_w, mdp, FF, D);

    dim3 gD(D / BN, T / BM);
    dim3 gH(1, T / BM);
    dim3 gF(FF / BN, T / BM);

    // 1) x = rmsnorm(hidden) -> f16
    rmsnorm_cast_kernel<<<T, blk, 0, stream>>>(hidden, norm1_w, x_h, D);
    // 2) projections (WMMA, branch-free)
    gemm_wmma<false><<<gD, blk, 0, stream>>>(x_h, qp,  qb,    nullptr, T, D, D, 0, 0);
    gemm_wmma<false><<<gD, blk, 0, stream>>>(x_h, kp,  kb,    nullptr, T, D, D, 0, 0);
    gemm_wmma<false><<<gD, blk, 0, stream>>>(x_h, vp,  vb,    nullptr, T, D, D, 0, 0);
    gemm_wmma<false><<<gD, blk, 0, stream>>>(x_h, gtp, gateb, nullptr, T, D, D, 0, 0);
    gemm_wmma<true ><<<gH, blk, 0, stream>>>(x_h, gp,  alphab, nullptr, T, H, D, 1, 0); // sigmoid
    gemm_wmma<true ><<<gH, blk, 0, stream>>>(x_h, bp,  betab,  nullptr, T, H, D, 1, 0); // sigmoid
    // 3) RoPE + L2-normalize q,k
    rope_l2_kernel<<<(T * H) / 8, blk, 0, stream>>>(qb, kb, T, H);
    // 4) sequential delta-rule scan
    delta_scan_kernel<<<H, blk, 0, stream>>>(qb, kb, vb, alphab, betab, ob, T, H);
    // 5) gated output norm -> f16
    opost_kernel<<<(T * H) / 8, blk, 0, stream>>>(ob, gateb, o_norm_w, o_h, T, H);
    // 6) h = hidden + o @ o_w (WMMA + fused residual)
    gemm_wmma<false><<<gD, blk, 0, stream>>>(o_h, op, hb, hidden, T, D, D, 0, 0);
    // 7) y = rmsnorm(h) -> f16
    rmsnorm_cast_kernel<<<T, blk, 0, stream>>>(hb, norm2_w, y_h, D);
    // 8) MLP: silu(y@Wg) * (y@Wu)  (WMMA, f16 outputs)
    gemm_wmma<false><<<gF, blk, 0, stream>>>(y_h, mgp, mg_h, nullptr, T, FF, D, 2, 1);
    gemm_wmma<false><<<gF, blk, 0, stream>>>(y_h, mup, mu_h, nullptr, T, FF, D, 0, 1);
    size_t nmul = (size_t)T * FF;
    mulcast_kernel<<<(unsigned)((nmul + 255) / 256), blk, 0, stream>>>(mg_h, mu_h, nmul);
    // 9) out = h + (silu*up) @ Wd  (WMMA + fused residual)
    gemm_wmma<false><<<gD, blk, 0, stream>>>(mg_h, mdp, (float*)d_out, hb, T, D, FF, 0, 0);
}
